// model_9620726743405
// MI455X (gfx1250) — matrix operand order repaired and checked by reading, not compiled, not run on hardware
//
#include <hip/hip_runtime.h>
#include <hip/hip_bf16.h>

typedef __attribute__((ext_vector_type(16))) _Float16 v16h;
typedef __attribute__((ext_vector_type(8)))  float    v8f;

#define LROWS     100   // items per set
#define L_PAD     128   // padded to 4 x K=32 WMMA chunks
#define DIMK      64
#define XS_STRIDE 72    // halves per row (64 + 8 pad, keeps 16B row alignment)
#define NEGF      (-1.0e30f)
#define EPSF      (1e-10f)

// Single LDS aggregate so async-to-LDS byte offsets are just offsetof()+index
// (the only LDS allocation in the kernel -> group-segment offset 0).
struct SLds {
  float    xsF[L_PAD * DIMK];       // f32 gather target (async loads land here)
  _Float16 xsH[L_PAD * XS_STRIDE];  // f16 staged xs for WMMA (zero-padded rows)
  float    attF[L_PAD];
  _Float16 wH[L_PAD];
  _Float16 aH[DIMK];
  int      Msh[L_PAD];
  float    sizesSh;
};

__device__ __forceinline__ float waveSum(float v) {
#pragma unroll
  for (int off = 16; off; off >>= 1) v += __shfl_xor(v, off, 32);
  return v;
}
__device__ __forceinline__ float waveMax(float v) {
#pragma unroll
  for (int off = 16; off; off >>= 1) v = fmaxf(v, __shfl_xor(v, off, 32));
  return v;
}
__device__ __forceinline__ int waveSumI(int v) {
#pragma unroll
  for (int off = 16; off; off >>= 1) v += __shfl_xor(v, off, 32);
  return v;
}

// masked softmax over 128 padded rows (executed by wave 0 only)
__device__ __forceinline__ void softmax128(const float* att, const int* msk,
                                           _Float16* wOut, int lane) {
  float v[4];
  float m = -3.4e38f;
#pragma unroll
  for (int i = 0; i < 4; ++i) {
    int l = lane + 32 * i;
    float a = (msk[l] > 0) ? att[l] : NEGF;
    v[i] = a;
    m = fmaxf(m, a);
  }
  m = waveMax(m);
  float s = 0.f;
#pragma unroll
  for (int i = 0; i < 4; ++i) { v[i] = __expf(v[i] - m); s += v[i]; }
  s = waveSum(s);
  float inv = 1.0f / s;
#pragma unroll
  for (int i = 0; i < 4; ++i) wOut[lane + 32 * i] = (_Float16)(v[i] * inv);
}

// D(16x16,f32) += A(16x32) * B(32x16).  A broadcasts w into ALL 16 rows
// (branch-free; rows are independent, we consume only row 0), B is an xs tile.
// Row 0 (lanes 0..15, element 0) holds (w^T xs)[nbase .. nbase+15].
__device__ __forceinline__ v8f wmma_vec_times_xs(const _Float16* w,
                                                 const _Float16* xs,
                                                 int nbase, int lane) {
  const int hi = lane >> 4;  // half-wave select: K 0..7/16..23 vs 8..15/24..31
  v8f acc = {};
#pragma unroll
  for (int kb = 0; kb < L_PAD; kb += 32) {
    const _Float16* wp = w + kb + 8 * hi;
    v16h aOp;
#pragma unroll
    for (int i = 0; i < 8; ++i) { aOp[i] = wp[i]; aOp[8 + i] = wp[16 + i]; }
    v16h bOp;  // B: lane = N col (nbase + lane%16), K in table order like A
    const _Float16* p = xs + (kb + 8 * hi) * XS_STRIDE + nbase + (lane & 15);
#pragma unroll
    for (int i = 0; i < 8; ++i) { bOp[i] = p[i * XS_STRIDE]; bOp[8 + i] = p[(16 + i) * XS_STRIDE]; }
    acc = __builtin_amdgcn_wmma_f32_16x16x32_f16(false, aOp, false, bOp,
                                                 (short)0, acc, false, false);
  }
  return acc;
}

__global__ __launch_bounds__(128) void attn_kernel(
    const int* __restrict__ S, const int* __restrict__ M,
    const float* __restrict__ X, const float* __restrict__ A,
    float* __restrict__ embOut, int size_reg) {
  __shared__ SLds s;

  const int wave = threadIdx.x >> 5;
  const int lane = threadIdx.x & 31;
  const int hi   = lane >> 4;
  const int set  = blockIdx.x;
  const long long sbase = (long long)set * LROWS;

  // per-lane attention vector coefficients (dims 2*lane, 2*lane+1)
  const float A0 = A[2 * lane], A1 = A[2 * lane + 1];

  // ---- Phase 1a: queue ALL row gathers straight into LDS (ASYNCcnt) ----
  for (int l = wave; l < LROWS; l += 4) {
    const int item = S[sbase + l];
    const unsigned ldsOff = (unsigned)__builtin_offsetof(SLds, xsF) +
                            (unsigned)((l * DIMK + 2 * lane) * 4);
    const unsigned long long ga =
        (unsigned long long)(const void*)(X + (size_t)item * DIMK + 2 * lane);
    asm volatile("global_load_async_to_lds_b64 %0, %1, off"
                 :: "v"(ldsOff), "v"(ga) : "memory");
  }
  asm volatile("s_wait_asynccnt 0" ::: "memory");  // this wave's rows landed

  // ---- Phase 1b: att = xs.A in f32 from LDS, down-convert xs to f16 ----
  for (int l = wave; l < L_PAD; l += 4) {
    float x0 = 0.f, x1 = 0.f;
    int mval = 0;
    if (l < LROWS) {
      const float2 xv = *(const float2*)&s.xsF[l * DIMK + 2 * lane];
      x0 = xv.x; x1 = xv.y;
      mval = M[sbase + l];
    }
    s.xsH[l * XS_STRIDE + 2 * lane]     = (_Float16)x0;
    s.xsH[l * XS_STRIDE + 2 * lane + 1] = (_Float16)x1;
    float att = waveSum(x0 * A0 + x1 * A1);
    if (lane == 0) { s.attF[l] = att; s.Msh[l] = mval; }
  }
  __syncthreads();

  // ---- Phase 2: masked softmax -> w (f16); set size ----
  if (wave == 0) {
    softmax128(s.attF, s.Msh, s.wH, lane);
    int cnt = 0;
#pragma unroll
    for (int i = 0; i < 4; ++i) cnt += s.Msh[lane + 32 * i];
    cnt = waveSumI(cnt);
    if (lane == 0) s.sizesSh = (float)cnt;
  }
  __syncthreads();

  // ---- Phase 3: a = w^T xs  (WMMA, one 16-col N tile per wave) ----
  {
    v8f acc = wmma_vec_times_xs(s.wH, s.xsH, wave * 16, lane);
    if (lane < 16) s.aH[wave * 16 + lane] = (_Float16)acc[0];
  }
  __syncthreads();

  // ---- Phase 4: att2 = xs @ a  (WMMA, 2 row tiles per wave) ----
#pragma unroll
  for (int t = 0; t < 2; ++t) {
    const int mbase = (wave * 2 + t) * 16;
    v8f acc = {};
#pragma unroll
    for (int kb = 0; kb < DIMK; kb += 32) {
      const int row = mbase + (lane & 15);
      const int k0  = kb + 8 * hi;
      const _Float16* p = s.xsH + row * XS_STRIDE;
      v16h aOp;
#pragma unroll
      for (int i = 0; i < 8; ++i) { aOp[i] = p[k0 + i]; aOp[8 + i] = p[k0 + 16 + i]; }
      const _Float16* ap = s.aH + kb + 8 * hi;  // a[k] (table K order), same for all 16 columns
      v16h bOp;
#pragma unroll
      for (int i = 0; i < 8; ++i) { bOp[i] = ap[i]; bOp[8 + i] = ap[16 + i]; }
      acc = __builtin_amdgcn_wmma_f32_16x16x32_f16(false, aOp, false, bOp,
                                                   (short)0, acc, false, false);
    }
    // every column of D equals att2; lanes 0 and 16 (one exec region) write
    if ((lane & 15) == 0) {
#pragma unroll
      for (int v = 0; v < 8; ++v) s.attF[mbase + 8 * hi + v] = acc[v];
    }
  }
  __syncthreads();

  // ---- Phase 5: second masked softmax -> w2 ----
  if (wave == 0) softmax128(s.attF, s.Msh, s.wH, lane);
  __syncthreads();

  // ---- Phase 6: emb = w2^T xs (WMMA) + size regularization, store ----
  {
    v8f acc = wmma_vec_times_xs(s.wH, s.xsH, wave * 16, lane);
    float scale = size_reg ? powf(s.sizesSh, 1.0f / 64.0f) : 1.0f;
    if (lane < 16)
      embOut[(size_t)set * DIMK + wave * 16 + lane] = acc[0] * scale;
  }
}

__device__ __forceinline__ float softplus1(float x) {
  // beta = 1: numerically stable log(1+exp(x))
  return fmaxf(x, 0.f) + log1pf(__expf(-fabsf(x)));
}

__global__ __launch_bounds__(256) void pair_kernel(
    const int* __restrict__ inst, const float* __restrict__ ov,
    const float* __restrict__ embc, const float* __restrict__ embr,
    float* __restrict__ out, int B) {
  const int wave = threadIdx.x >> 5;
  const int lane = threadIdx.x & 31;
  const int b = blockIdx.x * 8 + wave;
  if (b >= B) return;                 // whole wave exits together
  const int i = inst[2 * b], j = inst[2 * b + 1];
  const float2 ci = *(const float2*)(embc + (size_t)i * DIMK + 2 * lane);
  const float2 cj = *(const float2*)(embc + (size_t)j * DIMK + 2 * lane);
  const float2 ri = *(const float2*)(embr + (size_t)i * DIMK + 2 * lane);
  const float2 rj = *(const float2*)(embr + (size_t)j * DIMK + 2 * lane);

  float vol_i = 0.f, vol_j = 0.f, inter = 0.f, uni = 0.f;
#pragma unroll
  for (int c = 0; c < 2; ++c) {
    const float cI = c ? ci.y : ci.x, cJ = c ? cj.y : cj.x;
    const float rI = c ? ri.y : ri.x, rJ = c ? rj.y : rj.x;
    const float m_i = softplus1(cI), m_j = softplus1(cJ);
    const float e_i = softplus1(rI), e_j = softplus1(rJ);
    const float Mi = m_i + e_i, Mj = m_j + e_j;
    const float delta = fminf(Mi, Mj) - fmaxf(m_i, m_j);
    vol_i += logf(e_i + EPSF);
    vol_j += logf(e_j + EPSF);
    inter += logf(delta + EPSF);
    uni   += logf(fmaxf(Mi, Mj) - fminf(m_i, m_j) + EPSF);
  }
  vol_i = waveSum(vol_i);
  vol_j = waveSum(vol_j);
  inter = waveSum(inter);
  uni   = waveSum(uni);

  if (lane == 0) {
    float preds[4];
    preds[0] = inter;                                  // overlap
    preds[1] = inter / uni;                            // jaccard
    preds[2] = inter / powf(logf(vol_i + EPSF) + logf(vol_j + EPSF),
                            1.0f / 64.0f);             // cosine
    preds[3] = 2.f * inter / (vol_i + vol_j);          // dice
#pragma unroll
    for (int k = 0; k < 4; ++k) {
      const float d = __expf(preds[k]) - ov[(size_t)k * B + b];
      atomicAdd(&out[k], d * d);
    }
  }
}

__global__ void zero4(float* out) {
  if (threadIdx.x < 4) out[threadIdx.x] = 0.f;
}

extern "C" void kernel_launch(void* const* d_in, const int* in_sizes, int n_in,
                              void* d_out, int out_size, void* d_ws, size_t ws_size,
                              hipStream_t stream) {
  const int*   S    = (const int*)d_in[0];
  const int*   M    = (const int*)d_in[1];
  const int*   inst = (const int*)d_in[2];
  const float* ov   = (const float*)d_in[3];
  const float* Wc   = (const float*)d_in[4];
  const float* Wr   = (const float*)d_in[5];
  const float* Ac   = (const float*)d_in[6];
  const float* Ar   = (const float*)d_in[7];

  const int n_sets = in_sizes[0] / LROWS;
  const int B      = in_sizes[2] / 2;

  float* embc = (float*)d_ws;
  float* embr = embc + (size_t)n_sets * DIMK;
  float* out  = (float*)d_out;

  zero4<<<1, 64, 0, stream>>>(out);
  attn_kernel<<<n_sets, 128, 0, stream>>>(S, M, Wc, Ac, embc, 0);
  attn_kernel<<<n_sets, 128, 0, stream>>>(S, M, Wr, Ar, embr, 1);
  pair_kernel<<<(B + 7) / 8, 256, 0, stream>>>(inst, ov, embc, embr, out, B);
}
